// ConfigurableUNetGNN_3315714752656
// MI455X (gfx1250) — compile-verified, matrix operand order checked by reading, not run on hardware
//
#include <hip/hip_runtime.h>
#include <hip/hip_bf16.h>

typedef __attribute__((ext_vector_type(16))) _Float16 v16h;
typedef __attribute__((ext_vector_type(8)))  _Float16 v8h;
typedef __attribute__((ext_vector_type(8)))  float    v8f;

#define KNN_K 16

// ---------------------------------------------------------------------------
// Brute-force kNN (K=16) with register-resident insertion sort.
// ---------------------------------------------------------------------------
__global__ void knn16_kernel(const float* __restrict__ q, int nq,
                             const float* __restrict__ r, int nr,
                             int* __restrict__ out, int excl) {
    int i = blockIdx.x * blockDim.x + threadIdx.x;
    if (i >= nq) return;
    float qx = q[i * 3 + 0], qy = q[i * 3 + 1], qz = q[i * 3 + 2];
    float bd[KNN_K];
    int   bi[KNN_K];
#pragma unroll
    for (int t = 0; t < KNN_K; ++t) { bd[t] = __builtin_inff(); bi[t] = 0; }
    for (int j = 0; j < nr; ++j) {
        if (excl && j == i) continue;
        float dx = r[j * 3 + 0] - qx;
        float dy = r[j * 3 + 1] - qy;
        float dz = r[j * 3 + 2] - qz;
        float d = dx * dx + dy * dy + dz * dz;
        if (d < bd[KNN_K - 1]) {
            bd[KNN_K - 1] = d; bi[KNN_K - 1] = j;
#pragma unroll
            for (int t = KNN_K - 1; t > 0; --t) {
                if (bd[t] < bd[t - 1]) {
                    float td = bd[t]; bd[t] = bd[t - 1]; bd[t - 1] = td;
                    int   ti = bi[t]; bi[t] = bi[t - 1]; bi[t - 1] = ti;
                }
            }
        }
    }
#pragma unroll
    for (int t = 0; t < KNN_K; ++t) out[i * KNN_K + t] = bi[t];
}

// ---------------------------------------------------------------------------
// kNN k=3 for knn_interpolate; emits inverse-square-distance weights + sum.
// ---------------------------------------------------------------------------
__global__ void knn3_kernel(const float* __restrict__ q, int nq,
                            const float* __restrict__ r, int nr,
                            int* __restrict__ idx3, float* __restrict__ w3,
                            float* __restrict__ wsum) {
    int i = blockIdx.x * blockDim.x + threadIdx.x;
    if (i >= nq) return;
    float qx = q[i * 3 + 0], qy = q[i * 3 + 1], qz = q[i * 3 + 2];
    float d0 = __builtin_inff(), d1 = d0, d2 = d0;
    int i0 = 0, i1 = 0, i2 = 0;
    for (int j = 0; j < nr; ++j) {
        float dx = r[j * 3 + 0] - qx;
        float dy = r[j * 3 + 1] - qy;
        float dz = r[j * 3 + 2] - qz;
        float d = dx * dx + dy * dy + dz * dz;
        if (d < d2) {
            d2 = d; i2 = j;
            if (d2 < d1) { float t = d1; d1 = d2; d2 = t; int ti = i1; i1 = i2; i2 = ti; }
            if (d1 < d0) { float t = d0; d0 = d1; d1 = t; int ti = i0; i0 = i1; i1 = ti; }
        }
    }
    float w0 = 1.0f / (d0 + 1e-16f);
    float w1 = 1.0f / (d1 + 1e-16f);
    float w2 = 1.0f / (d2 + 1e-16f);
    idx3[i * 3 + 0] = i0; idx3[i * 3 + 1] = i1; idx3[i * 3 + 2] = i2;
    w3[i * 3 + 0] = w0;  w3[i * 3 + 1] = w1;  w3[i * 3 + 2] = w2;
    wsum[i] = w0 + w1 + w2;
}

// ---------------------------------------------------------------------------
// Farthest point sampling: single persistent block (inherently sequential).
// ---------------------------------------------------------------------------
__global__ void fps_kernel(const float* __restrict__ pos, int n, int m,
                           int* __restrict__ idxs, float* __restrict__ dmin) {
    __shared__ float sval[1024];
    __shared__ int   sidx[1024];
    __shared__ int   s_last;
    const int tid = threadIdx.x;
    const int bs  = blockDim.x;
    for (int i = tid; i < n; i += bs) dmin[i] = __builtin_inff();
    if (tid == 0) s_last = 0;
    __syncthreads();
    for (int it = 0; it < m; ++it) {
        int last = s_last;
        if (tid == 0) idxs[it] = last;
        float lx = pos[last * 3 + 0], ly = pos[last * 3 + 1], lz = pos[last * 3 + 2];
        float best = -1.0f; int bidx = 0x7fffffff;
        for (int i = tid; i < n; i += bs) {
            float dx = pos[i * 3 + 0] - lx;
            float dy = pos[i * 3 + 1] - ly;
            float dz = pos[i * 3 + 2] - lz;
            float d = dx * dx + dy * dy + dz * dz;
            float dm = fminf(dmin[i], d);
            dmin[i] = dm;
            if (dm > best || (dm == best && i < bidx)) { best = dm; bidx = i; }
        }
        sval[tid] = best; sidx[tid] = bidx;
        __syncthreads();
        for (int s = bs >> 1; s > 0; s >>= 1) {
            if (tid < s) {
                bool take = (sval[tid + s] > sval[tid]) ||
                            (sval[tid + s] == sval[tid] && sidx[tid + s] < sidx[tid]);
                if (take) { sval[tid] = sval[tid + s]; sidx[tid] = sidx[tid + s]; }
            }
            __syncthreads();
        }
        if (tid == 0) s_last = sidx[0];
        __syncthreads();
    }
}

// ---------------------------------------------------------------------------
// Row gather: dst[r,:] = src[idx[r],:]
// ---------------------------------------------------------------------------
__global__ void gather_rows(const float* __restrict__ src, const int* __restrict__ idx,
                            float* __restrict__ dst, int nrows, int C) {
    int t = blockIdx.x * blockDim.x + threadIdx.x;
    if (t >= nrows * C) return;
    int r = t / C, c = t - r * C;
    dst[t] = src[(size_t)idx[r] * C + c];
}

// ---------------------------------------------------------------------------
// knn_interpolate (k=3) fused with residual add.
// ---------------------------------------------------------------------------
__global__ void interp_apply(const float* __restrict__ xsrc, const int* __restrict__ idx3,
                             const float* __restrict__ w3, const float* __restrict__ wsum,
                             const float* __restrict__ rx, float* __restrict__ out,
                             int ny, int C) {
    int t = blockIdx.x * blockDim.x + threadIdx.x;
    if (t >= ny * C) return;
    int y = t / C, c = t - y * C;
    const int*   id = idx3 + y * 3;
    const float* ww = w3 + y * 3;
    float s = ww[0] * xsrc[(size_t)id[0] * C + c] +
              ww[1] * xsrc[(size_t)id[1] * C + c] +
              ww[2] * xsrc[(size_t)id[2] * C + c];
    out[t] = rx[t] + s / wsum[y];
}

// ---------------------------------------------------------------------------
// f32 -> f16 elementwise conversion (per-layer feature matrices).
// ---------------------------------------------------------------------------
__global__ void to_f16(const float* __restrict__ src, _Float16* __restrict__ dst, int n) {
    int t = blockIdx.x * blockDim.x + threadIdx.x;
    if (t < n) dst[t] = (_Float16)src[t];
}

// ---------------------------------------------------------------------------
// Repack W [2C, D] (f32, row-major) into per-lane WMMA B-fragment layout, f16:
//   Wh[ct][kt][lane][e]  (16 contiguous f16 per lane = one 32-byte load)
// For lane: column j = ct*16 + (lane&15), kbase = kt*32 + (lane>>4)*8,
// element e<8 -> K=kbase+e, e>=8 -> K=kbase+16+(e-8)   (per ISA A/B layout).
// blockIdx.y = ct; inner flat index tt enumerates (kt, lane, e).
// ---------------------------------------------------------------------------
__global__ void repack_w(const float* __restrict__ W, _Float16* __restrict__ Wh,
                         int twoC, int D) {
    const int ct = blockIdx.y;
    const int ktiles = twoC >> 5;
    int tt = blockIdx.x * blockDim.x + threadIdx.x;      // < ktiles*512
    if (tt >= (ktiles << 9)) return;
    int e    = tt & 15;
    int lane = (tt >> 4) & 31;
    int kt   = tt >> 9;
    int j     = (ct << 4) + (lane & 15);
    int kbase = (kt << 5) + ((lane >> 4) << 3);
    int k     = (e < 8) ? (kbase + e) : (kbase + 16 + (e - 8));
    Wh[((size_t)ct * (ktiles << 9)) + tt] = (_Float16)W[(size_t)k * D + j];
}

// ---------------------------------------------------------------------------
// EdgeConv via WMMA: out[p, j] = relu( max_k (h[p,k,:] @ W)[j] + b[j] )
//   h[p,k,c] = (c < C) ? x[p,c] : x[nbr[p,k], c-C] - x[p, c-C]
//
// Grid: (N/8 blocks of 8 waves, colTiles). One wave = one (point, 16-channel
// output tile). The 16 A-rows are the 16 neighbors of p, so the max-over-k is
// a max over 8 accumulator VGPRs + one shfl_xor(16) (f32 C-matrix layout).
// B fragments: pre-repacked f16 weights staged in LDS (shared by the block's
// 8 waves, which all use the same output-channel tile blockIdx.y).
// A fragments: aligned 16B packed-f16 loads; x_j - x_i via packed f16 sub.
// All control flow is wave-uniform, so EXEC is all-ones at every WMMA.
// ---------------------------------------------------------------------------
__global__ void __launch_bounds__(256)
edgeconv_wmma(const _Float16* __restrict__ xh, const int* __restrict__ nbr,
              const _Float16* __restrict__ Wh, const float* __restrict__ b,
              float* __restrict__ out, int N, int C, int D) {
    __shared__ __align__(32) _Float16 sW[16384];          // up to 32 K-tiles (32 KB)
    const int lane = threadIdx.x & 31;
    const int wave = threadIdx.x >> 5;
    const int ct   = blockIdx.y;
    const int ktiles = C >> 4;                            // (2C)/32

    // Stage this column-tile's W strip into LDS (ktiles * 1KB).
    {
        const uint4* src = (const uint4*)(Wh + (size_t)ct * (ktiles << 9));
        uint4*       dst = (uint4*)sW;
        const int total = ktiles << 6;                    // uint4 count
        for (int i = threadIdx.x; i < total; i += blockDim.x) dst[i] = src[i];
    }
    __syncthreads();

    const int p = blockIdx.x * (blockDim.x >> 5) + wave;
    if (p >= N) return;

    const int row = lane & 15;                 // A row (neighbor) == B column
    const int kh  = (lane >> 4) << 3;          // K sub-base per half-wave
    const int j   = (ct << 4) + row;           // output channel of this lane

    const _Float16* xph = xh + (size_t)p * C;
    const int       nb  = nbr[p * KNN_K + row];
    const _Float16* xnh = xh + (size_t)nb * C;

    v8f acc = {};
    for (int kt = 0; kt < ktiles; ++kt) {
        const int kbase = (kt << 5) + kh;
        const int s1 = kbase, s2 = kbase + 16;
        // Each 8-wide K-run lies entirely in the x_i half or the (x_j-x_i)
        // half (C, k0 are multiples of 32), and the predicate is wave-uniform.
        v8h r1, r2;
        if (s1 < C) r1 = *(const v8h*)(xph + s1);
        else        r1 = *(const v8h*)(xnh + (s1 - C)) - *(const v8h*)(xph + (s1 - C));
        if (s2 < C) r2 = *(const v8h*)(xph + s2);
        else        r2 = *(const v8h*)(xnh + (s2 - C)) - *(const v8h*)(xph + (s2 - C));
        v16h a = __builtin_shufflevector(r1, r2, 0, 1, 2, 3, 4, 5, 6, 7,
                                                 8, 9, 10, 11, 12, 13, 14, 15);
        const v16h bf = *(const v16h*)(sW + (((kt << 5) + lane) << 4));
        acc = __builtin_amdgcn_wmma_f32_16x16x32_f16(
            false, a, false, bf, (short)0, acc, false, false);
    }

    // max over 16 neighbor rows: 8 in this lane's accumulators, 8 in lane^16.
    float m = acc[0];
#pragma unroll
    for (int v = 1; v < 8; ++v) m = fmaxf(m, acc[v]);
    m = fmaxf(m, __shfl_xor(m, 16, 32));
    if (lane < 16) {
        float r = m + b[j];
        out[(size_t)p * D + j] = fmaxf(r, 0.0f);
    }
}

// ---------------------------------------------------------------------------
// Host orchestration
// ---------------------------------------------------------------------------
static inline int ceil_div(int a, int b) { return (a + b - 1) / b; }

static void run_edgeconv(const float* x, const int* nbr, const float* W,
                         const float* b, float* out, int N, int C, int D,
                         _Float16* xh, _Float16* Wh, hipStream_t s) {
    const int ne = N * C;
    to_f16<<<ceil_div(ne, 256), 256, 0, s>>>(x, xh, ne);
    const int twoC = 2 * C, ktiles = twoC >> 5, colTiles = D >> 4;
    dim3 rg(ceil_div(ktiles << 9, 256), colTiles);
    repack_w<<<rg, 256, 0, s>>>(W, Wh, twoC, D);
    dim3 g(ceil_div(N, 8), colTiles);
    edgeconv_wmma<<<g, 256, 0, s>>>(xh, nbr, Wh, b, out, N, C, D);
}

extern "C" void kernel_launch(void* const* d_in, const int* in_sizes, int n_in,
                              void* d_out, int out_size, void* d_ws, size_t ws_size,
                              hipStream_t stream) {
    (void)in_sizes; (void)n_in; (void)out_size; (void)ws_size;

    const float* x0   = (const float*)d_in[0];   // [16384, 64]
    const float* pos0 = (const float*)d_in[1];   // [16384, 3]
    const float* w_e0 = (const float*)d_in[2];  const float* b_e0 = (const float*)d_in[3];
    const float* w_e1 = (const float*)d_in[4];  const float* b_e1 = (const float*)d_in[5];
    const float* w_e2 = (const float*)d_in[6];  const float* b_e2 = (const float*)d_in[7];
    const float* w_e3 = (const float*)d_in[8];  const float* b_e3 = (const float*)d_in[9];
    const float* w_d0 = (const float*)d_in[10]; const float* b_d0 = (const float*)d_in[11];
    const float* w_d1 = (const float*)d_in[12]; const float* b_d1 = (const float*)d_in[13];
    const float* w_d2 = (const float*)d_in[14]; const float* b_d2 = (const float*)d_in[15];

    const int n0 = 16384, n1 = 4096, n2 = 1024, n3 = 256;

    char* wsp = (char*)d_ws;
    auto alloc = [&](size_t bytes) -> void* {
        void* p = (void*)wsp;
        wsp += (bytes + 255) & ~(size_t)255;
        return p;
    };

    int*   nbr0 = (int*)  alloc((size_t)n0 * KNN_K * 4);
    int*   nbr1 = (int*)  alloc((size_t)n1 * KNN_K * 4);
    int*   nbr2 = (int*)  alloc((size_t)n2 * KNN_K * 4);
    int*   nbr3 = (int*)  alloc((size_t)n3 * KNN_K * 4);
    float* e0   = (float*)alloc((size_t)n0 * 128 * 4);
    float* e1   = (float*)alloc((size_t)n1 * 256 * 4);
    float* e2   = (float*)alloc((size_t)n2 * 512 * 4);
    float* e3   = (float*)alloc((size_t)n3 * 512 * 4);
    float* pos1 = (float*)alloc((size_t)n1 * 3 * 4);
    float* pos2 = (float*)alloc((size_t)n2 * 3 * 4);
    float* pos3 = (float*)alloc((size_t)n3 * 3 * 4);
    int*   idx0 = (int*)  alloc((size_t)n1 * 4);
    int*   idx1 = (int*)  alloc((size_t)n2 * 4);
    int*   idx2 = (int*)  alloc((size_t)n3 * 4);
    float* dmin = (float*)alloc((size_t)n0 * 4);
    float* x1   = (float*)alloc((size_t)n1 * 128 * 4);
    float* x2   = (float*)alloc((size_t)n2 * 256 * 4);
    float* x3   = (float*)alloc((size_t)n3 * 512 * 4);
    float* xs0  = (float*)alloc((size_t)n2 * 512 * 4);
    float* d0o  = (float*)alloc((size_t)n2 * 256 * 4);
    float* xs1  = (float*)alloc((size_t)n1 * 256 * 4);
    float* d1o  = (float*)alloc((size_t)n1 * 128 * 4);
    float* xs2  = (float*)alloc((size_t)n0 * 128 * 4);
    int*   i3b  = (int*)  alloc((size_t)n0 * 3 * 4);
    float* w3b  = (float*)alloc((size_t)n0 * 3 * 4);
    float* wsm  = (float*)alloc((size_t)n0 * 4);
    _Float16* xh = (_Float16*)alloc((size_t)n0 * 128 * 2);   // reused per layer
    _Float16* Wh = (_Float16*)alloc((size_t)1024 * 512 * 2); // reused per layer

    const int TB = 256;

    // ---------------- Encoder ----------------
    knn16_kernel<<<ceil_div(n0, 128), 128, 0, stream>>>(pos0, n0, pos0, n0, nbr0, 1);
    run_edgeconv(x0, nbr0, w_e0, b_e0, e0, n0, 64, 128, xh, Wh, stream);
    fps_kernel<<<1, 1024, 0, stream>>>(pos0, n0, n1, idx0, dmin);
    gather_rows<<<ceil_div(n1 * 3, TB), TB, 0, stream>>>(pos0, idx0, pos1, n1, 3);
    gather_rows<<<ceil_div(n1 * 128, TB), TB, 0, stream>>>(e0, idx0, x1, n1, 128);

    knn16_kernel<<<ceil_div(n1, 128), 128, 0, stream>>>(pos1, n1, pos1, n1, nbr1, 1);
    run_edgeconv(x1, nbr1, w_e1, b_e1, e1, n1, 128, 256, xh, Wh, stream);
    fps_kernel<<<1, 1024, 0, stream>>>(pos1, n1, n2, idx1, dmin);
    gather_rows<<<ceil_div(n2 * 3, TB), TB, 0, stream>>>(pos1, idx1, pos2, n2, 3);
    gather_rows<<<ceil_div(n2 * 256, TB), TB, 0, stream>>>(e1, idx1, x2, n2, 256);

    knn16_kernel<<<ceil_div(n2, 128), 128, 0, stream>>>(pos2, n2, pos2, n2, nbr2, 1);
    run_edgeconv(x2, nbr2, w_e2, b_e2, e2, n2, 256, 512, xh, Wh, stream);
    fps_kernel<<<1, 1024, 0, stream>>>(pos2, n2, n3, idx2, dmin);
    gather_rows<<<ceil_div(n3 * 3, TB), TB, 0, stream>>>(pos2, idx2, pos3, n3, 3);
    gather_rows<<<ceil_div(n3 * 512, TB), TB, 0, stream>>>(e2, idx2, x3, n3, 512);

    knn16_kernel<<<ceil_div(n3, 128), 128, 0, stream>>>(pos3, n3, pos3, n3, nbr3, 1);
    run_edgeconv(x3, nbr3, w_e3, b_e3, e3, n3, 512, 512, xh, Wh, stream);

    // ---------------- Decoder ----------------
    knn3_kernel<<<ceil_div(n2, 128), 128, 0, stream>>>(pos2, n2, pos3, n3, i3b, w3b, wsm);
    interp_apply<<<ceil_div(n2 * 512, TB), TB, 0, stream>>>(e3, i3b, w3b, wsm, e2, xs0, n2, 512);
    run_edgeconv(xs0, nbr2, w_d0, b_d0, d0o, n2, 512, 256, xh, Wh, stream);

    knn3_kernel<<<ceil_div(n1, 128), 128, 0, stream>>>(pos1, n1, pos2, n2, i3b, w3b, wsm);
    interp_apply<<<ceil_div(n1 * 256, TB), TB, 0, stream>>>(d0o, i3b, w3b, wsm, e1, xs1, n1, 256);
    run_edgeconv(xs1, nbr1, w_d1, b_d1, d1o, n1, 256, 128, xh, Wh, stream);

    knn3_kernel<<<ceil_div(n0, 128), 128, 0, stream>>>(pos0, n0, pos1, n1, i3b, w3b, wsm);
    interp_apply<<<ceil_div(n0 * 128, TB), TB, 0, stream>>>(d1o, i3b, w3b, wsm, e0, xs2, n0, 128);
    run_edgeconv(xs2, nbr0, w_d2, b_d2, (float*)d_out, n0, 128, 64, xh, Wh, stream);
}
